// GAU_6262062317731
// MI455X (gfx1250) — matrix operand order repaired and checked by reading, not compiled, not run on hardware
//
#include <hip/hip_runtime.h>
#include <hip/hip_bf16.h>
#include <hip/hip_fp16.h>

// MI455X / gfx1250, wave32. All GEMMs on v_wmma_f32_16x16x32_f16.
// Register-blocked waves: 32x64 output per wave in the big GEMMs.

typedef __attribute__((ext_vector_type(16))) _Float16 v16h;
typedef __attribute__((ext_vector_type(8)))  float    v8f;

#define DIM   1024
#define QKD   128
#define HID   2048
#define BATCH 4
#define SEQ   2048
#define ROWS  (BATCH * SEQ)   // 8192
#define H2    (2 * HID)       // 4096

static __device__ __forceinline__ v8f wmma_f16f32(v16h a, v16h b, v8f c) {
  // D = A(16x32 f16) * B(32x16 f16) + C(16x16 f32)
  return __builtin_amdgcn_wmma_f32_16x16x32_f16(
      /*neg_a=*/false, a, /*neg_b=*/false, b,
      /*c_mod=*/(short)0, c, /*reuse_a=*/false, /*reuse_b=*/false);
}

// A fragment (16M x 32K) from row-major f16, base at tile origin, leading dim ld.
// Lanes 0-15 hold M=lane with K {0..7,16..23}; lanes 16-31 M=lane-16, K {8..15,24..31}.
static __device__ __forceinline__ v16h load_a_frag(const _Float16* base, int ld, int lane) {
  const int m  = lane & 15;
  const int kg = (lane >> 4) & 1;
  const _Float16* p = base + (size_t)m * ld + kg * 8;
  v16h a;
#pragma unroll
  for (int i = 0; i < 8; ++i) a[i] = p[i];          // K = kg*8 + i
#pragma unroll
  for (int i = 0; i < 8; ++i) a[8 + i] = p[16 + i]; // K = 16 + kg*8 + i
  return a;
}

// B fragment (32K x 16N); element (k,n) lives at baseT[n*ld + k] (N-major storage).
// Lane holds column N=lane&15; lanes 0-15 K {0..7,16..23}, lanes 16-31 K {8..15,24..31}.
static __device__ __forceinline__ v16h load_b_fragT(const _Float16* baseT, int ld, int lane) {
  const int n   = lane & 15;
  const int klo = ((lane >> 4) & 1) * 8;
  const _Float16* p = baseT + (size_t)n * ld + klo;
  v16h b;
#pragma unroll
  for (int i = 0; i < 16; ++i) b[i] = p[(i & 7) + ((i >> 3) << 4)];
  return b;
}

static __device__ __forceinline__ float silu_f(float v) {
  // x * rcp(1+e^-x): v_rcp_f32 instead of the IEEE divide expansion
  return v * __builtin_amdgcn_rcpf(1.0f + __expf(-v));
}

// ---------------- LayerNorm -> f16 ----------------
__global__ void ln_kernel(const float* __restrict__ x, const float* __restrict__ gamma,
                          const float* __restrict__ beta, _Float16* __restrict__ xn) {
  const int row = blockIdx.x;
  const float* xr = x + (size_t)row * DIM;
  float s = 0.f, s2 = 0.f;
  for (int i = threadIdx.x; i < DIM; i += blockDim.x) {
    float v = xr[i]; s += v; s2 += v * v;
  }
#pragma unroll
  for (int off = 16; off > 0; off >>= 1) {
    s  += __shfl_down(s, off, 32);
    s2 += __shfl_down(s2, off, 32);
  }
  __shared__ float red0[8], red1[8];
  __shared__ float mu_s, rstd_s;
  const int lane = threadIdx.x & 31, wave = threadIdx.x >> 5;
  if (lane == 0) { red0[wave] = s; red1[wave] = s2; }
  __syncthreads();
  if (threadIdx.x == 0) {
    float ts = 0.f, ts2 = 0.f;
    for (int w = 0; w < 8; ++w) { ts += red0[w]; ts2 += red1[w]; }
    float mu = ts / (float)DIM;
    float var = ts2 / (float)DIM - mu * mu;
    mu_s = mu; rstd_s = rsqrtf(var + 1e-5f);
  }
  __syncthreads();
  const float mu = mu_s, rstd = rstd_s;
  _Float16* xo = xn + (size_t)row * DIM;
  for (int i = threadIdx.x; i < DIM; i += blockDim.x) {
    float v = (xr[i] - mu) * rstd * gamma[i] + beta[i];
    xo[i] = (_Float16)v;
  }
}

// ---------------- f32 (KxN row-major) -> f16 transposed (NxK) ----------------
__global__ void convtr_kernel(const float* __restrict__ w, _Float16* __restrict__ wt,
                              int K, int N) {
  size_t idx = (size_t)blockIdx.x * blockDim.x + threadIdx.x;
  if (idx >= (size_t)K * N) return;
  int k = (int)(idx / N), n = (int)(idx % N);
  wt[(size_t)n * K + k] = (_Float16)w[idx];
}

// ---------------- hidden GEMM: xn @ W_hidden, SiLU, split v (transposed) / gate ----------------
// Wave owns 32x64 (2 A-frags x 4 B-frags, 8 accumulators). Block: 2x4 waves -> 64x256.
__global__ void hidden_gemm_kernel(const _Float16* __restrict__ xn, const _Float16* __restrict__ whT,
                                   const float* __restrict__ bh, _Float16* __restrict__ vt,
                                   _Float16* __restrict__ gate) {
  const int lane = threadIdx.x & 31, wave = threadIdx.x >> 5;
  const int m0 = blockIdx.y * 64 + (wave >> 2) * 32;
  const int n0 = blockIdx.x * 256 + (wave & 3) * 64;
  v8f acc[2][4] = {};
  for (int k0 = 0; k0 < DIM; k0 += 32) {
    if (k0 + 32 < DIM) {
      __builtin_prefetch(xn  + (size_t)(m0 + (lane & 15)) * DIM + k0 + 32, 0, 0);
      __builtin_prefetch(whT + (size_t)(n0 + (lane & 15)) * DIM + k0 + 32, 0, 0);
    }
    v16h af[2], bf[4];
#pragma unroll
    for (int i = 0; i < 2; ++i)
      af[i] = load_a_frag(xn + (size_t)(m0 + i * 16) * DIM + k0, DIM, lane);
#pragma unroll
    for (int j = 0; j < 4; ++j)
      bf[j] = load_b_fragT(whT + (size_t)(n0 + j * 16) * DIM + k0, DIM, lane);
#pragma unroll
    for (int i = 0; i < 2; ++i)
#pragma unroll
      for (int j = 0; j < 4; ++j)
        acc[i][j] = wmma_f16f32(af[i], bf[j], acc[i][j]);
  }
  const int n = lane & 15, mb = ((lane >> 4) & 1) * 8;
#pragma unroll
  for (int i = 0; i < 2; ++i) {
#pragma unroll
    for (int j = 0; j < 4; ++j) {
      const int col = n0 + j * 16 + n;
      const float bias = bh[col];
#pragma unroll
      for (int r = 0; r < 8; ++r) {
        const int row = m0 + i * 16 + mb + r;
        const float h = silu_f(acc[i][j][r] + bias);
        if (col < HID) { // v, stored d-major per batch: vt[b][d][j]
          const int b_ = row >> 11, jj = row & (SEQ - 1);
          vt[((size_t)b_ * HID + col) * SEQ + jj] = (_Float16)h;
        } else {         // gate, token-major
          gate[(size_t)row * HID + (col - HID)] = (_Float16)h;
        }
      }
    }
  }
}

// ---------------- qk GEMM: xn @ W_qk, SiLU, q/k via os_weight/os_bias ----------------
// Wave owns 32x16 (2 A-frags); 8 waves span all 128 qk columns.
__global__ void qk_gemm_kernel(const _Float16* __restrict__ xn, const _Float16* __restrict__ wqkT,
                               const float* __restrict__ bqk, const float* __restrict__ osw,
                               const float* __restrict__ osb,
                               _Float16* __restrict__ qh, _Float16* __restrict__ kh) {
  const int lane = threadIdx.x & 31, wave = threadIdx.x >> 5;
  const int m0 = blockIdx.x * 32;
  const int n0 = wave * 16; // 8 waves * 16 = 128 = QKD
  v8f acc[2] = {};
  for (int k0 = 0; k0 < DIM; k0 += 32) {
    v16h b = load_b_fragT(wqkT + (size_t)n0 * DIM + k0, DIM, lane);
#pragma unroll
    for (int i = 0; i < 2; ++i) {
      v16h a = load_a_frag(xn + (size_t)(m0 + i * 16) * DIM + k0, DIM, lane);
      acc[i] = wmma_f16f32(a, b, acc[i]);
    }
  }
  const int n = lane & 15, mb = ((lane >> 4) & 1) * 8;
  const int col = n0 + n;
  const float bias = bqk[col];
  const float wq = osw[col], wk = osw[QKD + col];
  const float bq = osb[col], bk = osb[QKD + col];
#pragma unroll
  for (int i = 0; i < 2; ++i) {
#pragma unroll
    for (int r = 0; r < 8; ++r) {
      const int row = m0 + i * 16 + mb + r;
      const float s = silu_f(acc[i][r] + bias);
      qh[(size_t)row * QKD + col] = (_Float16)(s * wq + bq);
      kh[(size_t)row * QKD + col] = (_Float16)(s * wk + bk);
    }
  }
}

// ---------------- attention: relu(q k^T / n)^2 @ v, fused gate (in-place) ----------------
__global__ void attn_kernel(const _Float16* __restrict__ qh, const _Float16* __restrict__ kh,
                            const _Float16* __restrict__ vt, _Float16* __restrict__ gate) {
  const int lane = threadIdx.x & 31, wave = threadIdx.x >> 5;
  const int b = blockIdx.x >> 7;            // 128 m-tiles per batch
  const int m0 = (blockIdx.x & 127) * 16;   // row tile within batch
  const int tok0 = b * SEQ + m0;
  const int d0 = blockIdx.y * 512 + wave * 64;

  __shared__ _Float16 sattn[8][16 * 32];    // per-wave 16x32 f16 score tile (LDS bounce)
  _Float16* smem = &sattn[wave][0];

  v16h qf[4];
#pragma unroll
  for (int t = 0; t < 4; ++t)
    qf[t] = load_a_frag(qh + (size_t)tok0 * QKD + t * 32, QKD, lane);

  v8f acc[4] = {};
  const float scale = 1.0f / (float)SEQ;
  const int n = lane & 15, mb = ((lane >> 4) & 1) * 8;
  const _Float16* kb = kh + (size_t)b * SEQ * QKD;
  const _Float16* vb = vt + (size_t)b * HID * SEQ;

  for (int j0 = 0; j0 < SEQ; j0 += 32) {
    v8f s0 = {}, s1 = {};
#pragma unroll
    for (int t = 0; t < 4; ++t) {
      v16h bk0 = load_b_fragT(kb + (size_t)j0 * QKD + t * 32, QKD, lane);
      s0 = wmma_f16f32(qf[t], bk0, s0);
      v16h bk1 = load_b_fragT(kb + (size_t)(j0 + 16) * QKD + t * 32, QKD, lane);
      s1 = wmma_f16f32(qf[t], bk1, s1);
    }
    // relu^2 epilogue; relayout C->A through per-wave LDS (same-wave DS ops are in order)
#pragma unroll
    for (int r = 0; r < 8; ++r) {
      const int m = mb + r;
      const float t0 = fmaxf(s0[r] * scale, 0.f);
      const float t1 = fmaxf(s1[r] * scale, 0.f);
      smem[m * 32 + n]      = (_Float16)(t0 * t0);
      smem[m * 32 + 16 + n] = (_Float16)(t1 * t1);
    }
    __asm volatile("s_wait_dscnt 0" ::: "memory");
    v16h af = load_a_frag(smem, 32, lane);
#pragma unroll
    for (int t = 0; t < 4; ++t) {
      v16h vv = load_b_fragT(vb + (size_t)(d0 + t * 16) * SEQ + j0, SEQ, lane);
      acc[t] = wmma_f16f32(af, vv, acc[t]);
    }
  }
  // out = (attn @ v) * gate ; overwrite gate buffer in place with gated f16
#pragma unroll
  for (int t = 0; t < 4; ++t) {
    const int col = d0 + t * 16 + n;
#pragma unroll
    for (int r = 0; r < 8; ++r) {
      const size_t idx = (size_t)(tok0 + mb + r) * HID + col;
      const float g = (float)gate[idx];
      gate[idx] = (_Float16)(acc[t][r] * g);
    }
  }
}

// ---------------- output GEMM: gated @ W_out + b_out -> f32 ----------------
// Wave owns 32x64; block: 2x4 waves -> 64x256.
__global__ void out_gemm_kernel(const _Float16* __restrict__ gated, const _Float16* __restrict__ woT,
                                const float* __restrict__ bo, float* __restrict__ out) {
  const int lane = threadIdx.x & 31, wave = threadIdx.x >> 5;
  const int m0 = blockIdx.y * 64 + (wave >> 2) * 32;
  const int n0 = blockIdx.x * 256 + (wave & 3) * 64;
  v8f acc[2][4] = {};
  for (int k0 = 0; k0 < HID; k0 += 32) {
    if (k0 + 32 < HID) {
      __builtin_prefetch(gated + (size_t)(m0 + (lane & 15)) * HID + k0 + 32, 0, 0);
      __builtin_prefetch(woT   + (size_t)(n0 + (lane & 15)) * HID + k0 + 32, 0, 0);
    }
    v16h af[2], bf[4];
#pragma unroll
    for (int i = 0; i < 2; ++i)
      af[i] = load_a_frag(gated + (size_t)(m0 + i * 16) * HID + k0, HID, lane);
#pragma unroll
    for (int j = 0; j < 4; ++j)
      bf[j] = load_b_fragT(woT + (size_t)(n0 + j * 16) * HID + k0, HID, lane);
#pragma unroll
    for (int i = 0; i < 2; ++i)
#pragma unroll
      for (int j = 0; j < 4; ++j)
        acc[i][j] = wmma_f16f32(af[i], bf[j], acc[i][j]);
  }
  const int n = lane & 15, mb = ((lane >> 4) & 1) * 8;
#pragma unroll
  for (int i = 0; i < 2; ++i) {
#pragma unroll
    for (int j = 0; j < 4; ++j) {
      const int col = n0 + j * 16 + n;
      const float bias = bo[col];
#pragma unroll
      for (int r = 0; r < 8; ++r)
        out[(size_t)(m0 + i * 16 + mb + r) * DIM + col] = acc[i][j][r] + bias;
    }
  }
}

extern "C" void kernel_launch(void* const* d_in, const int* in_sizes, int n_in,
                              void* d_out, int out_size, void* d_ws, size_t ws_size,
                              hipStream_t stream) {
  const float* x   = (const float*)d_in[0];
  const float* lns = (const float*)d_in[1];
  const float* lnb = (const float*)d_in[2];
  const float* wh  = (const float*)d_in[3];
  const float* bh  = (const float*)d_in[4];
  const float* wqk = (const float*)d_in[5];
  const float* bqk = (const float*)d_in[6];
  const float* osw = (const float*)d_in[7];
  const float* osb = (const float*)d_in[8];
  const float* wo  = (const float*)d_in[9];
  const float* bo  = (const float*)d_in[10];
  float* out = (float*)d_out;

  // Workspace carve-up (~96.3 MB total), 256B aligned.
  char* ws = (char*)d_ws;
  size_t off = 0;
  auto carve = [&](size_t bytes) {
    char* p = ws + off;
    off += (bytes + 255) & ~(size_t)255;
    return p;
  };
  _Float16* xn   = (_Float16*)carve((size_t)ROWS * DIM * 2);        // 16 MB
  _Float16* whT  = (_Float16*)carve((size_t)DIM * H2 * 2);          //  8 MB
  _Float16* wqkT = (_Float16*)carve((size_t)DIM * QKD * 2);         // .25 MB
  _Float16* woT  = (_Float16*)carve((size_t)HID * DIM * 2);         //  4 MB
  _Float16* vt   = (_Float16*)carve((size_t)BATCH * HID * SEQ * 2); // 32 MB
  _Float16* gate = (_Float16*)carve((size_t)ROWS * HID * 2);        // 32 MB (reused as gated)
  _Float16* qh   = (_Float16*)carve((size_t)ROWS * QKD * 2);        //  2 MB
  _Float16* kh   = (_Float16*)carve((size_t)ROWS * QKD * 2);        //  2 MB

  ln_kernel<<<ROWS, 256, 0, stream>>>(x, lns, lnb, xn);
  convtr_kernel<<<(DIM * H2) / 256, 256, 0, stream>>>(wh, whT, DIM, H2);
  convtr_kernel<<<(DIM * QKD) / 256, 256, 0, stream>>>(wqk, wqkT, DIM, QKD);
  convtr_kernel<<<(HID * DIM) / 256, 256, 0, stream>>>(wo, woT, HID, DIM);
  hidden_gemm_kernel<<<dim3(H2 / 256, ROWS / 64), 256, 0, stream>>>(xn, whT, bh, vt, gate);
  qk_gemm_kernel<<<ROWS / 32, 256, 0, stream>>>(xn, wqkT, bqk, osw, osb, qh, kh);
  attn_kernel<<<dim3(BATCH * (SEQ / 16), HID / 512), 256, 0, stream>>>(qh, kh, vt, gate);
  out_gemm_kernel<<<dim3(DIM / 256, ROWS / 64), 256, 0, stream>>>(gate, woT, bo, out);
}
